// GridNetBlock_attnhead_74466142978183
// MI455X (gfx1250) — matrix operand order repaired and checked by reading, not compiled, not run on hardware
//
#include <hip/hip_runtime.h>
#include <hip/hip_bf16.h>
#include <math.h>

// ---------------------------------------------------------------------------
// GridNet attention block (4 layers) for MI455X / gfx1250.
// All matmuls use v_wmma_f32_16x16x32_bf16 (wave32 WMMA, f32 accumulate).
// ---------------------------------------------------------------------------

typedef __attribute__((ext_vector_type(16))) __bf16 v16bf;
typedef __attribute__((ext_vector_type(8)))  float  v8f;

// problem dims
constexpr int NB   = 4;          // batch
constexpr int NC   = 64;         // emb channels
constexpr int NT   = 800;        // time (400 pos + 400 neg)
constexpr int NF   = 65;         // freqs
constexpr int NTF  = NT * NF;    // 52000
constexpr int NH   = 4;          // heads
constexpr int NE   = 8;          // q/k channels per head
constexpr int NCV  = 16;         // v channels per head
constexpr int DQK  = NE * NF;    // 520
constexpr int DQKP = 544;        // padded to multiple of 32 (WMMA K)
constexpr int DV   = NCV * NF;   // 1040
constexpr int LAYERS = 4;

// GEMM tile config: 256 threads = 8 waves; each wave does 32(M) x 64(N).
constexpr int BM = 128, BN = 128, BK = 32, BKP = 40; // BKP pads LDS rows (80B)

enum GMode { G_QKV = 0, G_SCORES = 1, G_PV = 2, G_PROJ = 3 };

union FragU { uint4 u[2]; v16bf v; };

// ------------------------------ reductions ---------------------------------
__device__ __forceinline__ float waveSum(float v) {
#pragma unroll
  for (int o = 16; o > 0; o >>= 1) v += __shfl_xor(v, o, 32);
  return v;
}
__device__ __forceinline__ float waveMax(float v) {
#pragma unroll
  for (int o = 16; o > 0; o >>= 1) v = fmaxf(v, __shfl_xor(v, o, 32));
  return v;
}
__device__ __forceinline__ float blockSum(float v, float* sm) {
  int lane = threadIdx.x & 31, wid = threadIdx.x >> 5;
  v = waveSum(v);
  if (lane == 0) sm[wid] = v;
  __syncthreads();
  float r = (threadIdx.x < 8) ? sm[threadIdx.x] : 0.f;
  if (wid == 0) { r = waveSum(r); if (lane == 0) sm[0] = r; }
  __syncthreads();
  r = sm[0];
  __syncthreads();
  return r;
}
__device__ __forceinline__ float blockMax(float v, float* sm) {
  int lane = threadIdx.x & 31, wid = threadIdx.x >> 5;
  v = waveMax(v);
  if (lane == 0) sm[wid] = v;
  __syncthreads();
  float r = (threadIdx.x < 8) ? sm[threadIdx.x] : -3.4e38f;
  if (wid == 0) { r = waveMax(r); if (lane == 0) sm[0] = r; }
  __syncthreads();
  r = sm[0];
  __syncthreads();
  return r;
}

// ------------------------- generic bf16 WMMA GEMM --------------------------
// C[M,N] = A[M,K](bf16,row-major) * B, f32 accumulate.
//  G_QKV / G_PROJ : B source is f32 [K, ldb] (k-major), converted to bf16
//                   while staging into LDS (transposed to [n][k]).
//  G_SCORES/G_PV  : B source is bf16 [N, ldb] row-major over K ("B^T" form),
//                   so Q*K^T and P*V both read contiguous rows.
//  G_PV epilogue scatters O[t, cv*65+f] -> O[b, h*16+cv, t, f].
template <int MODE>
__global__ __launch_bounds__(256) void gemm_kernel(
    const __bf16* __restrict__ A, const void* __restrict__ Bsrc,
    float* __restrict__ C, int M, int N, int K, int lda, int ldb, int ldc,
    long sA, long sB, long sC, float scale) {
  __shared__ __bf16 ldsA[BM * BKP];
  __shared__ __bf16 ldsB[BN * BKP];

  const int tid   = threadIdx.x;
  const int lane  = tid & 31;
  const int wid   = tid >> 5;
  const int waveM = wid >> 1;  // 0..3  -> 32 rows each
  const int waveN = wid & 1;   // 0..1  -> 64 cols each
  const int half  = lane >> 4; // WMMA lane-half
  const int l15   = lane & 15;

  const int m0 = blockIdx.y * BM;
  const int n0 = blockIdx.x * BN;
  const int batch = blockIdx.z;

  const __bf16* Ab = A + (long)batch * sA;

  v8f acc[2][4];
#pragma unroll
  for (int i = 0; i < 2; ++i)
#pragma unroll
    for (int j = 0; j < 4; ++j)
#pragma unroll
      for (int r = 0; r < 8; ++r) acc[i][j][r] = 0.f;

  const int ktiles = K / BK;  // K is always a multiple of 32 here
  for (int kt = 0; kt < ktiles; ++kt) {
    const int k0 = kt * BK;

    // ---- stage global -> registers ----
    uint4 a0 = {0, 0, 0, 0}, a1 = {0, 0, 0, 0};
    uint4 b0 = {0, 0, 0, 0}, b1 = {0, 0, 0, 0};
    float bv[16];
    {
      int row = tid >> 1, h2 = tid & 1;
      int gm = m0 + row;
      if (gm < M) {
        const uint4* p = (const uint4*)(Ab + (long)gm * lda + k0 + h2 * 16);
        a0 = p[0]; a1 = p[1];
        __builtin_prefetch(Ab + (long)gm * lda + k0 + BK, 0, 1);
      }
    }
    if constexpr (MODE == G_QKV || MODE == G_PROJ) {
      const float* Bf = (const float*)Bsrc + (long)batch * sB;
      int k = tid >> 3;       // 0..31
      int chunk = tid & 7;    // 16 n's each
      int nb = n0 + chunk * 16;
      const float* p = Bf + (long)(k0 + k) * ldb + nb;
      if (nb + 16 <= N) {
        const float4* p4 = (const float4*)p;
        float4 f0 = p4[0], f1 = p4[1], f2 = p4[2], f3 = p4[3];
        bv[0] = f0.x; bv[1] = f0.y; bv[2]  = f0.z; bv[3]  = f0.w;
        bv[4] = f1.x; bv[5] = f1.y; bv[6]  = f1.z; bv[7]  = f1.w;
        bv[8] = f2.x; bv[9] = f2.y; bv[10] = f2.z; bv[11] = f2.w;
        bv[12] = f3.x; bv[13] = f3.y; bv[14] = f3.z; bv[15] = f3.w;
      } else {
#pragma unroll
        for (int e = 0; e < 16; ++e) bv[e] = (nb + e < N) ? p[e] : 0.f;
      }
    } else {
      const __bf16* Bh = (const __bf16*)Bsrc + (long)batch * sB;
      int row = tid >> 1, h2 = tid & 1;
      int gn = n0 + row;
      if (gn < N) {
        const uint4* p = (const uint4*)(Bh + (long)gn * ldb + k0 + h2 * 16);
        b0 = p[0]; b1 = p[1];
      }
    }

    __syncthreads();
    // ---- registers -> LDS ----
    {
      int row = tid >> 1, h2 = tid & 1;
      uint4* dst = (uint4*)&ldsA[row * BKP + h2 * 16];
      dst[0] = a0; dst[1] = a1;
    }
    if constexpr (MODE == G_QKV || MODE == G_PROJ) {
      int k = tid >> 3, chunk = tid & 7;
#pragma unroll
      for (int e = 0; e < 16; ++e)
        ldsB[(chunk * 16 + e) * BKP + k] = (__bf16)bv[e];
    } else {
      int row = tid >> 1, h2 = tid & 1;
      uint4* dst = (uint4*)&ldsB[row * BKP + h2 * 16];
      dst[0] = b0; dst[1] = b1;
    }
    __syncthreads();

    // ---- build fragments per documented CDNA5 WMMA layouts ----
    FragU af[2], bfr[4];
#pragma unroll
    for (int mi = 0; mi < 2; ++mi) {
      // A 16x32 bf16: lane l -> M=l&15; elems 0..7 -> K=half*8+e,
      //               elems 8..15 -> K=16+half*8+(e-8)
      const __bf16* base = &ldsA[(waveM * 32 + mi * 16 + l15) * BKP];
      af[mi].u[0] = *(const uint4*)(base + half * 8);
      af[mi].u[1] = *(const uint4*)(base + 16 + half * 8);
    }
#pragma unroll
    for (int ni = 0; ni < 4; ++ni) {
      // B 32x16 bf16: lane l -> N=l&15; elems 0..7 -> K=half*8+e,
      //               elems 8..15 -> K=16+half*8+(e-8)
      const __bf16* base = &ldsB[(waveN * 64 + ni * 16 + l15) * BKP];
      bfr[ni].u[0] = *(const uint4*)(base + half * 8);
      bfr[ni].u[1] = *(const uint4*)(base + 16 + half * 8);
    }
#pragma unroll
    for (int mi = 0; mi < 2; ++mi)
#pragma unroll
      for (int ni = 0; ni < 4; ++ni)
        acc[mi][ni] = __builtin_amdgcn_wmma_f32_16x16x32_bf16(
            false, af[mi].v, false, bfr[ni].v, (short)0, acc[mi][ni], false,
            false);
  }

  // ---- epilogue: C/D layout M = r + 8*half, N = lane&15 ----
#pragma unroll
  for (int mi = 0; mi < 2; ++mi)
#pragma unroll
    for (int ni = 0; ni < 4; ++ni)
#pragma unroll
      for (int r = 0; r < 8; ++r) {
        int m = m0 + waveM * 32 + mi * 16 + r + half * 8;
        int n = n0 + waveN * 64 + ni * 16 + l15;
        if (m < M && n < N) {
          float v = acc[mi][ni][r] * scale;
          if constexpr (MODE == G_PV) {
            int h = batch >> 2, b = batch & 3;       // batch = h*NB + b
            int cv = n / 65, f = n - cv * 65;
            C[((long)(b * NC + h * NCV + cv) * NT + m) * NF + f] = v;
          } else {
            C[(long)batch * sC + (long)m * ldc + n] = v;
          }
        }
      }
}

// ------------------------ elementwise / LN kernels -------------------------

// x = concat(pos,neg, dim=t) + seg_table[t<400?0:1]
__global__ __launch_bounds__(256) void init_x_kernel(
    const float* __restrict__ pos, const float* __restrict__ neg,
    const float* __restrict__ seg, float* __restrict__ x0) {
  long idx = (long)blockIdx.x * 256 + threadIdx.x;  // [b][c][t*65+f]
  int n = (int)(idx % NTF);
  int bc = (int)(idx / NTF);
  int c = bc & 63, b = bc >> 6;
  int t = n / 65, f = n - t * 65;
  float v;
  if (t < 400)
    v = pos[((long)(b * NC + c) * 400 + t) * 65 + f] + seg[c * 65 + f];
  else
    v = neg[((long)(b * NC + c) * 400 + (t - 400)) * 65 + f] +
        seg[NC * NF + c * 65 + f];
  x0[idx] = v;
}

// pack per-layer QKV weights [128,64] and proj weights [64,64] to bf16
__global__ __launch_bounds__(256) void pack_w_kernel(
    const float* __restrict__ Qw_l, const float* __restrict__ Kw_l,
    const float* __restrict__ Vw_l, const float* __restrict__ Pw_l,
    __bf16* __restrict__ Wc, __bf16* __restrict__ Pwb) {
  int i = blockIdx.x * 256 + threadIdx.x;
  if (i < 128 * 64) {
    int ch = i >> 6, c = i & 63;
    float v;
    if (ch < 32)      v = Qw_l[ch * 64 + c];          // ch = h*8+e
    else if (ch < 64) v = Kw_l[(ch - 32) * 64 + c];
    else              v = Vw_l[(ch - 64) * 64 + c];   // ch-64 = h*16+cv
    Wc[i] = (__bf16)v;
  } else if (i < 128 * 64 + 64 * 64) {
    int j = i - 128 * 64;
    Pwb[j] = (__bf16)Pw_l[j];
  }
}

// PReLU + LayerNorm over (channel,freq) per (h,b,t); writes bf16 Q/K rows
// (K-padded to 544) and V transposed [hb][d][t].
__global__ __launch_bounds__(256) void ln_qkv_kernel(
    const float* __restrict__ raw,  // [b][128][52000]
    const float* __restrict__ Qbias, const float* __restrict__ Kbias,
    const float* __restrict__ Vbias, const float* __restrict__ Qa,
    const float* __restrict__ Ka, const float* __restrict__ Va,
    const float* __restrict__ Qg, const float* __restrict__ Qbe,
    const float* __restrict__ Kg, const float* __restrict__ Kbe,
    const float* __restrict__ Vg, const float* __restrict__ Vbe,
    __bf16* __restrict__ Qh, __bf16* __restrict__ Kh,
    __bf16* __restrict__ Vh) {
  __shared__ float sm[8];
  int sel = blockIdx.y;           // 0=Q 1=K 2=V
  int t = blockIdx.x % NT;
  int hb = blockIdx.x / NT;       // h*NB + b
  int h = hb >> 2, b = hb & 3;
  int nO = (sel == 2) ? NCV : NE;
  int count = nO * NF;
  int ch0 = (sel == 0) ? h * NE : (sel == 1) ? 32 + h * NE : 64 + h * NCV;
  const float* bias = (sel == 0) ? Qbias + h * NE
                      : (sel == 1) ? Kbias + h * NE : Vbias + h * NCV;
  float alpha = (sel == 0) ? Qa[h] : (sel == 1) ? Ka[h] : Va[h];
  const float* g  = (sel == 0) ? Qg + h * NE * NF
                    : (sel == 1) ? Kg + h * NE * NF : Vg + h * NCV * NF;
  const float* be = (sel == 0) ? Qbe + h * NE * NF
                    : (sel == 1) ? Kbe + h * NE * NF : Vbe + h * NCV * NF;

  int tid = threadIdx.x;
  float loc[5];
  float s = 0.f, ss = 0.f;
  for (int i = tid, j = 0; i < count; i += 256, ++j) {
    int o = i / 65, f = i - o * 65;
    float v = raw[(long)(b * 128 + ch0 + o) * NTF + t * 65 + f] + bias[o];
    v = v >= 0.f ? v : alpha * v;
    loc[j] = v; s += v; ss += v * v;
  }
  s = blockSum(s, sm);
  ss = blockSum(ss, sm);
  float mu = s / count;
  float rstd = rsqrtf(ss / count - mu * mu + 1e-5f);
  for (int i = tid, j = 0; i < count; i += 256, ++j) {
    float v = (loc[j] - mu) * rstd * g[i] + be[i];
    if (sel == 2) {
      Vh[((long)hb * DV + i) * NT + t] = (__bf16)v;      // transposed [d][t]
    } else {
      __bf16* row = ((sel == 0) ? Qh : Kh) + ((long)hb * NT + t) * DQKP;
      row[i] = (__bf16)v;
    }
  }
  if (sel != 2) {  // zero the K-pad 520..543
    __bf16* row = ((sel == 0) ? Qh : Kh) + ((long)hb * NT + t) * DQKP;
    for (int i = DQK + tid; i < DQKP; i += 256) row[i] = (__bf16)0.f;
  }
}

// row softmax over s, write bf16 probs
__global__ __launch_bounds__(256) void softmax_kernel(
    const float* __restrict__ S, __bf16* __restrict__ P) {
  __shared__ float sm[8];
  int t = blockIdx.x, hb = blockIdx.y;
  const float* row = S + ((long)hb * NT + t) * NT;
  __bf16* prow = P + ((long)hb * NT + t) * NT;
  int tid = threadIdx.x;
  float loc[4];
  float mx = -3.4e38f;
  for (int i = tid, j = 0; i < NT; i += 256, ++j) {
    loc[j] = row[i];
    mx = fmaxf(mx, loc[j]);
  }
  mx = blockMax(mx, sm);
  float s = 0.f;
  for (int i = tid, j = 0; i < NT; i += 256, ++j) {
    loc[j] = __expf(loc[j] - mx);
    s += loc[j];
  }
  s = blockSum(s, sm);
  float inv = 1.f / s;
  for (int i = tid, j = 0; i < NT; i += 256, ++j)
    prow[i] = (__bf16)(loc[j] * inv);
}

// PReLU + LayerNorm over (channel,freq) per (b,t) + optional residual
__global__ __launch_bounds__(256) void ln_proj_kernel(
    const float* __restrict__ Yraw, const float* __restrict__ Pbias,
    const float* __restrict__ Pa_l, const float* __restrict__ Pg_l,
    const float* __restrict__ Pbe_l, const float* __restrict__ xin,
    float* __restrict__ xout, int addResidual) {
  __shared__ float sm[8];
  int t = blockIdx.x, b = blockIdx.y;
  const int count = NC * NF;  // 4160
  float alpha = Pa_l[0];
  int tid = threadIdx.x;
  float loc[17];
  float s = 0.f, ss = 0.f;
  for (int i = tid, j = 0; i < count; i += 256, ++j) {
    int c = i / 65, f = i - c * 65;
    float v = Yraw[(long)(b * NC + c) * NTF + t * 65 + f] + Pbias[c];
    v = v >= 0.f ? v : alpha * v;
    loc[j] = v; s += v; ss += v * v;
  }
  s = blockSum(s, sm);
  ss = blockSum(ss, sm);
  float mu = s / count;
  float rstd = rsqrtf(ss / count - mu * mu + 1e-5f);
  for (int i = tid, j = 0; i < count; i += 256, ++j) {
    int c = i / 65, f = i - c * 65;
    float v = (loc[j] - mu) * rstd * Pg_l[i] + Pbe_l[i];
    long idx = (long)(b * NC + c) * NTF + t * 65 + f;
    if (addResidual) v += xin[idx];
    xout[idx] = v;
  }
}

// ------------------------------- host side ---------------------------------
extern "C" void kernel_launch(void* const* d_in, const int* in_sizes, int n_in,
                              void* d_out, int out_size, void* d_ws,
                              size_t ws_size, hipStream_t stream) {
  (void)in_sizes; (void)n_in; (void)out_size; (void)ws_size;
  const float* pos = (const float*)d_in[0];
  const float* neg = (const float*)d_in[1];
  const float* seg = (const float*)d_in[2];
  const float* Qw  = (const float*)d_in[3];
  const float* Qbi = (const float*)d_in[4];
  const float* Qa  = (const float*)d_in[5];
  const float* Qg  = (const float*)d_in[6];
  const float* Qbe = (const float*)d_in[7];
  const float* Kw  = (const float*)d_in[8];
  const float* Kbi = (const float*)d_in[9];
  const float* Ka  = (const float*)d_in[10];
  const float* Kg  = (const float*)d_in[11];
  const float* Kbe = (const float*)d_in[12];
  const float* Vw  = (const float*)d_in[13];
  const float* Vbi = (const float*)d_in[14];
  const float* Va  = (const float*)d_in[15];
  const float* Vg  = (const float*)d_in[16];
  const float* Vbe = (const float*)d_in[17];
  const float* Pw  = (const float*)d_in[18];
  const float* Pbi = (const float*)d_in[19];
  const float* Pa  = (const float*)d_in[20];
  const float* Pg  = (const float*)d_in[21];
  const float* Pbe = (const float*)d_in[22];

  char* ws = (char*)d_ws;
  size_t off = 0;
  auto alloc = [&](size_t bytes) -> void* {
    void* p = ws + off;
    off = (off + bytes + 255) & ~(size_t)255;
    return p;
  };
  float*  x0   = (float*)alloc(sizeof(float) * (size_t)NB * NC * NTF);
  float*  x1   = (float*)alloc(sizeof(float) * (size_t)NB * NC * NTF);
  float*  qkvr = (float*)alloc(sizeof(float) * (size_t)NB * 128 * NTF);
  __bf16* Qh   = (__bf16*)alloc(2ull * NH * NB * NT * DQKP);
  __bf16* Kh   = (__bf16*)alloc(2ull * NH * NB * NT * DQKP);
  __bf16* Vh   = (__bf16*)alloc(2ull * NH * NB * DV * NT);
  float*  S    = (float*)alloc(4ull * NH * NB * NT * NT);
  __bf16* Pp   = (__bf16*)alloc(2ull * NH * NB * NT * NT);
  float*  O    = (float*)alloc(sizeof(float) * (size_t)NB * NC * NTF);
  float*  Yr   = (float*)alloc(sizeof(float) * (size_t)NB * NC * NTF);
  __bf16* Wc   = (__bf16*)alloc(2ull * 128 * 64);
  __bf16* Pwb  = (__bf16*)alloc(2ull * 64 * 64);

  // x = concat + segment embedding
  init_x_kernel<<<(NB * NC * NTF) / 256, 256, 0, stream>>>(pos, neg, seg, x0);

  const float invsq = 1.0f / sqrtf((float)DQK);
  float* xc = x0;
  float* xn = x1;
  for (int li = 0; li < LAYERS; ++li) {
    pack_w_kernel<<<48, 256, 0, stream>>>(
        Qw + (size_t)li * NH * NE * NC, Kw + (size_t)li * NH * NE * NC,
        Vw + (size_t)li * NH * NCV * NC, Pw + (size_t)li * NC * NC, Wc, Pwb);

    // fused QKV projection: [128,64] x [64, 52000] per batch
    dim3 g1((NTF + BN - 1) / BN, 1, NB);
    gemm_kernel<G_QKV><<<g1, 256, 0, stream>>>(
        Wc, xc, qkvr, 128, NTF, 64, 64, NTF, NTF, 0L, (long)NC * NTF,
        128L * NTF, 1.0f);

    dim3 g2(NH * NB * NT, 3);
    ln_qkv_kernel<<<g2, 256, 0, stream>>>(
        qkvr, Qbi + (size_t)li * NH * NE, Kbi + (size_t)li * NH * NE,
        Vbi + (size_t)li * NH * NCV, Qa + (size_t)li * NH,
        Ka + (size_t)li * NH, Va + (size_t)li * NH,
        Qg + (size_t)li * NH * NE * NF, Qbe + (size_t)li * NH * NE * NF,
        Kg + (size_t)li * NH * NE * NF, Kbe + (size_t)li * NH * NE * NF,
        Vg + (size_t)li * NH * NCV * NF, Vbe + (size_t)li * NH * NCV * NF,
        Qh, Kh, Vh);

    // scores = Q K^T / sqrt(d)
    dim3 g3((NT + BN - 1) / BN, (NT + BM - 1) / BM, NH * NB);
    gemm_kernel<G_SCORES><<<g3, 256, 0, stream>>>(
        Qh, Kh, S, NT, NT, DQKP, DQKP, DQKP, NT, (long)NT * DQKP,
        (long)NT * DQKP, (long)NT * NT, invsq);

    dim3 g4(NT, NH * NB);
    softmax_kernel<<<g4, 256, 0, stream>>>(S, Pp);

    // O = P V  (epilogue scatters to concat-head layout)
    dim3 g5((DV + BN - 1) / BN, (NT + BM - 1) / BM, NH * NB);
    gemm_kernel<G_PV><<<g5, 256, 0, stream>>>(
        Pp, Vh, O, NT, DV, NT, NT, NT, 0, (long)NT * NT, (long)DV * NT, 0L,
        1.0f);

    // concat projection: [64,64] x [64, 52000]
    dim3 g6((NTF + BN - 1) / BN, 1, NB);
    gemm_kernel<G_PROJ><<<g6, 256, 0, stream>>>(
        Pwb, O, Yr, NC, NTF, NC, 64, NTF, NTF, 0L, (long)NC * NTF,
        (long)NC * NTF, 1.0f);

    float* dst = (li == LAYERS - 1) ? (float*)d_out : xn;
    dim3 g7(NT, NB);
    ln_proj_kernel<<<g7, 256, 0, stream>>>(
        Yr, Pbi + (size_t)li * NC, Pa + li, Pg + (size_t)li * NC * NF,
        Pbe + (size_t)li * NC * NF, xc, dst, (li == 0) ? 1 : 0);

    float* oldxc = xc;
    xc = dst;
    xn = oldxc;
  }
}
